// ConstrainedDecoderForSplitDec_7421703487623
// MI455X (gfx1250) — compile-verified, matrix operand order checked by reading, not run on hardware
//
#include <hip/hip_runtime.h>
#include <hip/hip_bf16.h>
#include <math.h>

// ---------------- problem constants ----------------
namespace {
constexpr int B_    = 256;
constexpr int S_    = 256;
constexpr int T_    = 48;
constexpr int EMB_  = 256;
constexpr int HID_  = 1024;
constexpr int NLAB_ = 128;
constexpr int KGATES = 2 * EMB_ + HID_;   // 1536  (xcat ++ h)
constexpr int NGATES = 4 * HID_;          // 4096
constexpr int KHO    = 2 * HID_;          // 2048  (weighted ++ h_lstm)
constexpr int KOUT   = 2 * HID_ + EMB_;   // 2304  (hidden ++ weighted ++ x)

typedef __attribute__((ext_vector_type(16))) __bf16 v16bf;
typedef __attribute__((ext_vector_type(8)))  __bf16 v8bf;
typedef __attribute__((ext_vector_type(4)))  __bf16 v4bf;
typedef __attribute__((ext_vector_type(8)))  float  v8f;
} // namespace

// CDNA5 has a hardware V_TANH_F32 transcendental (TRANS op, co-executes with
// VALU). Prefer the gfx1250 builtin; fall back safely if the name differs.
#if __has_builtin(__builtin_amdgcn_tanhf)
#define FAST_TANH(x) __builtin_amdgcn_tanhf(x)
#elif __has_builtin(__builtin_amdgcn_tanh_f32)
#define FAST_TANH(x) __builtin_amdgcn_tanh_f32(x)
#else
#define FAST_TANH(x) tanhf(x)
#endif

// sigmoid via hardware tanh: sigma(x) = 0.5 + 0.5*tanh(x/2)
static __device__ __forceinline__ float fast_sigmoid(float x) {
    return 0.5f + 0.5f * FAST_TANH(0.5f * x);
}

// Load one 16x32 bf16 B-fragment (column tile) for this lane.
// Lane layout per CDNA5 ISA: lanes 0-15 hold K {0..7,16..23} of col n0+lane,
// lanes 16-31 hold K {8..15,24..31} of col n0+lane-16.
static __device__ __forceinline__ v16bf load_bfrag(const __bf16* base) {
    v8bf b0 = *(const v8bf*)(base);
    v8bf b1 = *(const v8bf*)(base + 16);
    v16bf bv;
#pragma unroll
    for (int i = 0; i < 8; ++i) {
        bv[i]     = b0[i];
        bv[i + 8] = b1[i];
    }
    return bv;
}

// ---------------------------------------------------------------------------
// WMMA GEMM:  C[M,N] = act(A[M,K](f32) * Bt[N,K](bf16, pre-transposed) + bias)
// Each wave owns a 16(M) x 64(N) tile: ONE A-fragment is reused by FOUR
// v_wmma_f32_16x16x32_bf16 ops per K-step (4x less A traffic, 4 WMMAs in
// flight). Block = 256 threads = 8 waves arranged 4(M) x 2(N) -> 64x128 tile.
// A is converted f32->bf16 in registers (v_cvt_pk_bf16_f32), f32 accumulate.
// D layout: M = reg + 8*(lane>>4), N = lane&15 (per CDNA5 ISA).
// ---------------------------------------------------------------------------
template <int ACT, bool OUT_BF16>
__global__ void wmma_gemm_f32a(const float* __restrict__ A,
                               const __bf16* __restrict__ Bt,
                               const float* __restrict__ bias,
                               void* __restrict__ Cout,
                               int M, int N, int K, int ldc) {
    const int wave = threadIdx.x >> 5;
    const int lane = threadIdx.x & 31;
    const int m0 = (blockIdx.y * 4 + (wave >> 1)) * 16;
    const int n0 = (blockIdx.x * 2 + (wave & 1)) * 64;
    if (m0 >= M || n0 >= N) return;

    const int lm     = lane & 15;
    const int kb_off = (lane >> 4) << 3;      // 0 or 8

    const float*  arow = A  + (size_t)(m0 + lm) * K;
    const __bf16* brow = Bt + (size_t)(n0 + lm) * K;
    const size_t  bstp = (size_t)16 * K;      // next 16-column group

    v8f acc0 = {}, acc1 = {}, acc2 = {}, acc3 = {};
    for (int k0 = 0; k0 < K; k0 += 32) {
        const int kb = k0 + kb_off;
        v8f a0 = *(const v8f*)(arow + kb);
        v8f a1 = *(const v8f*)(arow + kb + 16);
        v16bf av;
#pragma unroll
        for (int i = 0; i < 8; ++i) {
            av[i]     = (__bf16)a0[i];
            av[i + 8] = (__bf16)a1[i];
        }
        v16bf bv0 = load_bfrag(brow + kb);
        v16bf bv1 = load_bfrag(brow + bstp + kb);
        v16bf bv2 = load_bfrag(brow + 2 * bstp + kb);
        v16bf bv3 = load_bfrag(brow + 3 * bstp + kb);
        acc0 = __builtin_amdgcn_wmma_f32_16x16x32_bf16(false, av, false, bv0,
                                                       (short)0, acc0, false, false);
        acc1 = __builtin_amdgcn_wmma_f32_16x16x32_bf16(false, av, false, bv1,
                                                       (short)0, acc1, false, false);
        acc2 = __builtin_amdgcn_wmma_f32_16x16x32_bf16(false, av, false, bv2,
                                                       (short)0, acc2, false, false);
        acc3 = __builtin_amdgcn_wmma_f32_16x16x32_bf16(false, av, false, bv3,
                                                       (short)0, acc3, false, false);
    }

    const int mbase = m0 + ((lane >> 4) << 3);
    v8f accs[4] = {acc0, acc1, acc2, acc3};
#pragma unroll
    for (int j = 0; j < 4; ++j) {
        const int   ncol = n0 + j * 16 + lm;
        const float bval = bias ? bias[ncol] : 0.0f;
#pragma unroll
        for (int r = 0; r < 8; ++r) {
            float v = accs[j][r] + bval;
            if (ACT == 1) v = FAST_TANH(v);
            const size_t idx = (size_t)(mbase + r) * ldc + ncol;
            if (OUT_BF16) ((__bf16*)Cout)[idx] = (__bf16)v;
            else          ((float*)Cout)[idx]  = v;
        }
    }
}

// ---------------------------------------------------------------------------
// Convert f32 weight [K,N] (row-major) to bf16 transposed [N, ldout] at kofs.
// ---------------------------------------------------------------------------
__global__ void conv_wT(const float* __restrict__ in, __bf16* __restrict__ out,
                        int K, int N, int ldout, int kofs) {
    const int idx = blockIdx.x * 256 + threadIdx.x;
    if (idx >= K * N) return;
    const int k = idx / N, n = idx % N;
    out[(size_t)n * ldout + kofs + k] = (__bf16)in[idx];
}

__global__ void vec_add2(const float* a, const float* b, float* o, int n) {
    const int i = blockIdx.x * 256 + threadIdx.x;
    if (i < n) o[i] = a[i] + b[i];
}

__global__ void vec_copy(const float* a, float* o, int n) {
    const int i = blockIdx.x * 256 + threadIdx.x;
    if (i < n) o[i] = a[i];
}

// ---------------------------------------------------------------------------
// xcat[b,t,:] = [ x_t(256) , emb(labels[b,t])(256) ]
// x_0 = decoder_input[b];  x_t = emb(cur_labels[b,t-1]) for t>=1.
// ---------------------------------------------------------------------------
__global__ void build_xcat(const float* __restrict__ dec_in,
                           const float* __restrict__ emb,
                           const int* __restrict__ cur_labels,
                           const int* __restrict__ labels,
                           float* __restrict__ xcat) {
    const int idx = blockIdx.x * 256 + threadIdx.x;
    const int total = B_ * T_ * (2 * EMB_);
    if (idx >= total) return;
    const int col = idx % (2 * EMB_);
    const int t   = (idx / (2 * EMB_)) % T_;
    const int b   = idx / (2 * EMB_ * T_);
    float v;
    if (col < EMB_) {
        if (t == 0) v = dec_in[(size_t)b * EMB_ + col];
        else        v = emb[(size_t)cur_labels[b * T_ + t - 1] * EMB_ + col];
    } else {
        v = emb[(size_t)labels[b * T_ + t] * EMB_ + (col - EMB_)];
    }
    xcat[idx] = v;
}

// ---------------------------------------------------------------------------
// dst[b, 0:w0] = s0[b*ld0 ..], dst[b, w0:w0+w1] = s1, dst[b, ..] = s2
// ---------------------------------------------------------------------------
__global__ void concat3(float* __restrict__ dst, int W,
                        const float* __restrict__ s0, int w0, int ld0,
                        const float* __restrict__ s1, int w1, int ld1,
                        const float* __restrict__ s2, int w2, int ld2) {
    const int idx = blockIdx.x * 256 + threadIdx.x;
    if (idx >= B_ * W) return;
    const int b = idx / W, col = idx % W;
    float v;
    if (col < w0)           v = s0[(size_t)b * ld0 + col];
    else if (col < w0 + w1) v = s1[(size_t)b * ld1 + (col - w0)];
    else                    v = s2[(size_t)b * ld2 + (col - w0 - w1)];
    dst[idx] = v;
}

// ---------------------------------------------------------------------------
// LSTM elementwise: gates [B, 4H] (i,f,g,o) ; c updated in place; h_lstm out.
// Sigmoids/tanh use the CDNA5 hardware tanh transcendental.
// ---------------------------------------------------------------------------
__global__ void lstm_elem(const float* __restrict__ gates,
                          float* __restrict__ c, float* __restrict__ h_lstm) {
    const int idx = blockIdx.x * 256 + threadIdx.x;
    if (idx >= B_ * HID_) return;
    const int b = idx / HID_, h = idx % HID_;
    const float* g = gates + (size_t)b * NGATES;
    const float ig = g[h];
    const float fg = g[HID_ + h];
    const float gg = g[2 * HID_ + h];
    const float og = g[3 * HID_ + h];
    const float si = fast_sigmoid(ig);
    const float sf = fast_sigmoid(fg);
    const float so = fast_sigmoid(og);
    const float cn = sf * c[idx] + si * FAST_TANH(gg);
    c[idx]      = cn;
    h_lstm[idx] = so * FAST_TANH(cn);
}

// ---------------------------------------------------------------------------
// Bahdanau attention for one batch row per block (256 threads, 8 waves).
//   e[s]     = sum_h V[h] * tanh(q[b,h] + ctxp[b,s,h])
//   alpha    = softmax(e)
//   weighted = sum_s alpha[s] * ctxp[b,s,:]
// ctxp is bf16 (128 MB total -> resident in the 192 MB L2 across all 48 steps).
// Pass 1 keeps q/V slices in registers and loads ctx rows as 128-bit v8bf;
// pass 2 uses 64-bit v4bf loads (wave reads 256B contiguous per s).
// ---------------------------------------------------------------------------
__global__ void attn_kernel(const float* __restrict__ q,
                            const __bf16* __restrict__ ctxp,
                            const float* __restrict__ V,
                            float* __restrict__ weighted) {
    const int b = blockIdx.x;
    __shared__ float se[S_];
    __shared__ float redA[8];
    __shared__ float redB[8];

    const int tid  = threadIdx.x;
    const int wave = tid >> 5;
    const int lane = tid & 31;

    // Per-lane register copies of q[b,:] and V slices: h = j*256 + lane*8 + i
    float sqr[4][8], svr[4][8];
#pragma unroll
    for (int j = 0; j < 4; ++j) {
        v8f qv = *(const v8f*)(q + (size_t)b * HID_ + j * 256 + lane * 8);
        v8f vv = *(const v8f*)(V + j * 256 + lane * 8);
#pragma unroll
        for (int i = 0; i < 8; ++i) { sqr[j][i] = qv[i]; svr[j][i] = vv[i]; }
    }

    const __bf16* cb = ctxp + (size_t)b * S_ * HID_;

    // pass 1: scores (each wave owns s = wave, wave+8, ...)
    for (int s = wave; s < S_; s += 8) {
        const __bf16* row = cb + (size_t)s * HID_ + lane * 8;
        float p = 0.0f;
#pragma unroll
        for (int j = 0; j < 4; ++j) {
            v8bf rv = *(const v8bf*)(row + j * 256);
#pragma unroll
            for (int i = 0; i < 8; ++i)
                p += svr[j][i] * FAST_TANH(sqr[j][i] + (float)rv[i]);
        }
        for (int off = 16; off > 0; off >>= 1)
            p += __shfl_down(p, off, 32);
        if (lane == 0) se[s] = p;
    }
    __syncthreads();

    // softmax over se[0..S-1] (one element per thread)
    float e = se[tid];
    float m = e;
    for (int off = 16; off > 0; off >>= 1)
        m = fmaxf(m, __shfl_down(m, off, 32));
    if (lane == 0) redA[wave] = m;
    __syncthreads();
    if (tid == 0) {
        float mm = redA[0];
        for (int i = 1; i < 8; ++i) mm = fmaxf(mm, redA[i]);
        redA[0] = mm;
    }
    __syncthreads();
    const float gmax = redA[0];
    const float ex   = __expf(e - gmax);
    float ssum = ex;
    for (int off = 16; off > 0; off >>= 1)
        ssum += __shfl_down(ssum, off, 32);
    if (lane == 0) redB[wave] = ssum;
    __syncthreads();
    if (tid == 0) {
        float t = 0.0f;
        for (int i = 0; i < 8; ++i) t += redB[i];
        redB[0] = t;
    }
    __syncthreads();
    se[tid] = ex / redB[0];       // alpha
    __syncthreads();

    // pass 2: weighted context; each thread owns 4 consecutive h (64-bit loads)
    float a0 = 0.0f, a1 = 0.0f, a2 = 0.0f, a3 = 0.0f;
    const __bf16* colp = cb + tid * 4;
    for (int s = 0; s < S_; ++s) {
        v4bf rv = *(const v4bf*)(colp + (size_t)s * HID_);
        const float al = se[s];
        a0 += al * (float)rv[0];
        a1 += al * (float)rv[1];
        a2 += al * (float)rv[2];
        a3 += al * (float)rv[3];
    }
    float* wp = weighted + (size_t)b * HID_ + tid * 4;
    wp[0] = a0; wp[1] = a1; wp[2] = a2; wp[3] = a3;
}

// ---------------------------------------------------------------------------
__global__ void argmax_kernel(const float* __restrict__ outs,
                              int* __restrict__ preds) {
    const int idx = blockIdx.x * 256 + threadIdx.x;   // over B*T
    if (idx >= B_ * T_) return;
    const float* r = outs + (size_t)idx * NLAB_;
    float best = r[0];
    int   bi   = 0;
    for (int n = 1; n < NLAB_; ++n) {
        const float v = r[n];
        if (v > best) { best = v; bi = n; }
    }
    preds[idx] = bi;
}

// ===========================================================================
extern "C" void kernel_launch(void* const* d_in, const int* in_sizes, int n_in,
                              void* d_out, int out_size, void* d_ws, size_t ws_size,
                              hipStream_t stream) {
    (void)in_sizes; (void)n_in; (void)out_size; (void)ws_size;

    const float* decoder_input = (const float*)d_in[2];
    const float* h0        = (const float*)d_in[3];
    const float* c0        = (const float*)d_in[4];
    const float* context   = (const float*)d_in[5];
    const int*   cur_labels= (const int*)d_in[6];
    const int*   labels    = (const int*)d_in[7];
    const float* emb_table = (const float*)d_in[8];
    const float* W_ih      = (const float*)d_in[9];
    const float* b_ih      = (const float*)d_in[10];
    const float* W_hh      = (const float*)d_in[11];
    const float* b_hh      = (const float*)d_in[12];
    const float* W_ho      = (const float*)d_in[13];
    const float* b_ho      = (const float*)d_in[14];
    const float* W_ain     = (const float*)d_in[15];
    const float* b_ain     = (const float*)d_in[16];
    const float* W_actx    = (const float*)d_in[17];
    const float* b_actx    = (const float*)d_in[18];
    const float* Vv        = (const float*)d_in[19];
    const float* W_out     = (const float*)d_in[20];
    const float* b_out     = (const float*)d_in[21];

    // ---- workspace carve (256B aligned) ----
    char* p = (char*)d_ws;
    auto alloc = [&](size_t bytes) -> void* {
        void* r = (void*)p;
        p += (bytes + 255) & ~(size_t)255;
        return r;
    };
    __bf16* ctxp      = (__bf16*)alloc((size_t)B_ * S_ * HID_ * 2);     // 128 MB
    float*  xcat      = (float*) alloc((size_t)B_ * T_ * 2 * EMB_ * 4); // 25 MB
    __bf16* Wt_comb   = (__bf16*)alloc((size_t)NGATES * KGATES * 2);
    __bf16* Wt_actx   = (__bf16*)alloc((size_t)HID_ * HID_ * 2);
    __bf16* Wt_ain    = (__bf16*)alloc((size_t)HID_ * HID_ * 2);
    __bf16* Wt_ho     = (__bf16*)alloc((size_t)HID_ * KHO * 2);
    __bf16* Wt_out    = (__bf16*)alloc((size_t)NLAB_ * KOUT * 2);
    float*  bias_comb = (float*) alloc((size_t)NGATES * 4);
    float*  hbuf      = (float*) alloc((size_t)B_ * HID_ * 4);
    float*  cbuf      = (float*) alloc((size_t)B_ * HID_ * 4);
    float*  gates     = (float*) alloc((size_t)B_ * NGATES * 4);
    float*  h_lstm    = (float*) alloc((size_t)B_ * HID_ * 4);
    float*  qbuf      = (float*) alloc((size_t)B_ * HID_ * 4);
    float*  wtd       = (float*) alloc((size_t)B_ * HID_ * 4);
    float*  Acat1     = (float*) alloc((size_t)B_ * KGATES * 4);
    float*  Acat2     = (float*) alloc((size_t)B_ * KHO * 4);
    float*  Acat3     = (float*) alloc((size_t)B_ * KOUT * 4);

    float* outF  = (float*)d_out;                       // [B,T,NLAB]
    int*   preds = (int*)(outF + (size_t)B_ * T_ * NLAB_);
    float* hf    = (float*)(preds + (size_t)B_ * T_);
    float* cf    = hf + (size_t)B_ * HID_;

    auto gemm = [&](const float* A, const __bf16* Bt, const float* bias,
                    void* C, int M, int N, int K, int ldc, int act, bool obf) {
        dim3 grid(N / 128, M / 64);
        if (obf)
            wmma_gemm_f32a<0, true><<<grid, 256, 0, stream>>>(A, Bt, bias, C, M, N, K, ldc);
        else if (act == 1)
            wmma_gemm_f32a<1, false><<<grid, 256, 0, stream>>>(A, Bt, bias, C, M, N, K, ldc);
        else
            wmma_gemm_f32a<0, false><<<grid, 256, 0, stream>>>(A, Bt, bias, C, M, N, K, ldc);
    };

    // ---- one-time prep ----
    conv_wT<<<(1024 * 1024 + 255) / 256, 256, 0, stream>>>(W_actx, Wt_actx, HID_, HID_, HID_, 0);
    conv_wT<<<(1024 * 1024 + 255) / 256, 256, 0, stream>>>(W_ain,  Wt_ain,  HID_, HID_, HID_, 0);
    conv_wT<<<(2048 * 1024 + 255) / 256, 256, 0, stream>>>(W_ho,   Wt_ho,   KHO,  HID_, KHO,  0);
    conv_wT<<<(2304 * 128 + 255) / 256, 256, 0, stream>>>(W_out,  Wt_out,  KOUT, NLAB_, KOUT, 0);
    conv_wT<<<(512 * 4096 + 255) / 256, 256, 0, stream>>>(W_ih, Wt_comb, 2 * EMB_, NGATES, KGATES, 0);
    conv_wT<<<(1024 * 4096 + 255) / 256, 256, 0, stream>>>(W_hh, Wt_comb, HID_, NGATES, KGATES, 2 * EMB_);
    vec_add2<<<(NGATES + 255) / 256, 256, 0, stream>>>(b_ih, b_hh, bias_comb, NGATES);
    build_xcat<<<(B_ * T_ * 2 * EMB_ + 255) / 256, 256, 0, stream>>>(
        decoder_input, emb_table, cur_labels, labels, xcat);
    vec_copy<<<(B_ * HID_ + 255) / 256, 256, 0, stream>>>(h0, hbuf, B_ * HID_);
    vec_copy<<<(B_ * HID_ + 255) / 256, 256, 0, stream>>>(c0, cbuf, B_ * HID_);

    // ctx_proj (bf16 out, stays hot in L2 for all 48 attention passes)
    gemm(context, Wt_actx, b_actx, ctxp, B_ * S_, HID_, HID_, HID_, 0, true);

    // ---- recurrent loop ----
    const int ld_xcat = T_ * 2 * EMB_;
    for (int t = 0; t < T_; ++t) {
        const float* xcat_t = xcat + (size_t)t * 2 * EMB_;

        // Acat1 = [x_t(512) , h(1024)]
        concat3<<<(B_ * KGATES + 255) / 256, 256, 0, stream>>>(
            Acat1, KGATES, xcat_t, 2 * EMB_, ld_xcat, hbuf, HID_, HID_, hbuf, 0, HID_);
        // gates = Acat1 @ [W_ih;W_hh] + (b_ih + b_hh)
        gemm(Acat1, Wt_comb, bias_comb, gates, B_, NGATES, KGATES, NGATES, 0, false);
        lstm_elem<<<(B_ * HID_ + 255) / 256, 256, 0, stream>>>(gates, cbuf, h_lstm);

        // q = h_lstm @ W_ain + b_ain
        gemm(h_lstm, Wt_ain, b_ain, qbuf, B_, HID_, HID_, HID_, 0, false);
        // attention: scores + softmax + weighted context
        attn_kernel<<<B_, 256, 0, stream>>>(qbuf, ctxp, Vv, wtd);

        // hidden_t = tanh([weighted, h_lstm] @ W_ho + b_ho)  -> becomes new h
        concat3<<<(B_ * KHO + 255) / 256, 256, 0, stream>>>(
            Acat2, KHO, wtd, HID_, HID_, h_lstm, HID_, HID_, h_lstm, 0, HID_);
        gemm(Acat2, Wt_ho, b_ho, hbuf, B_, HID_, KHO, HID_, 1, false);

        // out = [hidden, weighted, x_t] @ W_out + b_out  -> d_out[:, t, :]
        concat3<<<(B_ * KOUT + 255) / 256, 256, 0, stream>>>(
            Acat3, KOUT, hbuf, HID_, HID_, wtd, HID_, HID_, xcat_t, EMB_, ld_xcat);
        gemm(Acat3, Wt_out, b_out, outF + (size_t)t * NLAB_,
             B_, NLAB_, KOUT, T_ * NLAB_, 0, false);
    }

    // ---- finishers ----
    argmax_kernel<<<(B_ * T_ + 255) / 256, 256, 0, stream>>>(outF, preds);
    vec_copy<<<(B_ * HID_ + 255) / 256, 256, 0, stream>>>(hbuf, hf, B_ * HID_);
    vec_copy<<<(B_ * HID_ + 255) / 256, 256, 0, stream>>>(cbuf, cf, B_ * HID_);
}
